// GCNModel_43731357008191
// MI455X (gfx1250) — compile-verified, matrix operand order checked by reading, not run on hardware
//
#include <hip/hip_runtime.h>

#define NN 100000
#define NE 1600000
#define DIN 128
#define DHID 64
#define BN_EPS 1e-5f

typedef __attribute__((ext_vector_type(16))) __bf16 v16bf;
typedef __attribute__((ext_vector_type(8)))  float  v8f;

// ---------------- W1 pre-swizzle into per-lane WMMA B layout ----------------
// Bp group index t = nt*128 + kc*32 + lane ; 16 bf16 per group (32B).
// Lane layout for 16-bit B (K x N, 32x16): lane = half*16 + n,
//   elems 0..7  = B[kb + 0..7 , col],  elems 8..15 = B[kb+16+0..7, col]
//   with kb = kc*32 + half*8, col = nt*16 + n.
__global__ void k_pack_w(const float* __restrict__ W1, __bf16* __restrict__ Bp) {
    int t = blockIdx.x * blockDim.x + threadIdx.x;
    if (t >= 512) return;
    int lane = t & 31;
    int kc   = (t >> 5) & 3;
    int nt   = t >> 7;
    int n    = lane & 15;
    int half = lane >> 4;
    int col  = nt * 16 + n;
    int kb   = kc * 32 + half * 8;
    __bf16* dst = Bp + t * 16;
#pragma unroll
    for (int j = 0; j < 8; ++j) dst[j]     = (__bf16)W1[(kb + j) * DHID + col];
#pragma unroll
    for (int j = 0; j < 8; ++j) dst[8 + j] = (__bf16)W1[(kb + 16 + j) * DHID + col];
}

// ---------------- h0 = x @ W1  (one wave -> 16 rows x 64 cols) ----------------
__global__ void k_gemm(const float* __restrict__ x, const __bf16* __restrict__ Bp,
                       float* __restrict__ h0) {
    int wid  = (blockIdx.x * blockDim.x + threadIdx.x) >> 5;
    int lane = threadIdx.x & 31;
    int row0 = wid * 16;
    if (row0 >= NN) return;
    int m    = lane & 15;
    int half = lane >> 4;
    const float* arow = x + (size_t)(row0 + m) * DIN;
    const v16bf* bmat = (const v16bf*)Bp;

    v8f acc0 = {}, acc1 = {}, acc2 = {}, acc3 = {};
#pragma unroll
    for (int kc = 0; kc < 4; ++kc) {
        int kb = kc * 32 + half * 8;
        float4 f0 = *(const float4*)(arow + kb);
        float4 f1 = *(const float4*)(arow + kb + 4);
        float4 f2 = *(const float4*)(arow + kb + 16);
        float4 f3 = *(const float4*)(arow + kb + 20);
        v16bf a;
        a[0]  = (__bf16)f0.x; a[1]  = (__bf16)f0.y; a[2]  = (__bf16)f0.z; a[3]  = (__bf16)f0.w;
        a[4]  = (__bf16)f1.x; a[5]  = (__bf16)f1.y; a[6]  = (__bf16)f1.z; a[7]  = (__bf16)f1.w;
        a[8]  = (__bf16)f2.x; a[9]  = (__bf16)f2.y; a[10] = (__bf16)f2.z; a[11] = (__bf16)f2.w;
        a[12] = (__bf16)f3.x; a[13] = (__bf16)f3.y; a[14] = (__bf16)f3.z; a[15] = (__bf16)f3.w;

        v16bf b0 = bmat[(0 * 4 + kc) * 32 + lane];
        v16bf b1 = bmat[(1 * 4 + kc) * 32 + lane];
        v16bf b2 = bmat[(2 * 4 + kc) * 32 + lane];
        v16bf b3 = bmat[(3 * 4 + kc) * 32 + lane];
        acc0 = __builtin_amdgcn_wmma_f32_16x16x32_bf16(false, a, false, b0, (short)0, acc0, false, false);
        acc1 = __builtin_amdgcn_wmma_f32_16x16x32_bf16(false, a, false, b1, (short)0, acc1, false, false);
        acc2 = __builtin_amdgcn_wmma_f32_16x16x32_bf16(false, a, false, b2, (short)0, acc2, false, false);
        acc3 = __builtin_amdgcn_wmma_f32_16x16x32_bf16(false, a, false, b3, (short)0, acc3, false, false);
    }
    // C/D layout: VGPR r -> row = r + 8*half, col = lane&15 (+ 16*nt)
#pragma unroll
    for (int r = 0; r < 8; ++r) {
        float* orow = h0 + (size_t)(row0 + r + 8 * half) * DHID + m;
        orow[0]  = acc0[r];
        orow[16] = acc1[r];
        orow[32] = acc2[r];
        orow[48] = acc3[r];
    }
}

// ---------------- zero agg, deg = 1 (self loop) ----------------
__global__ void k_init(float* __restrict__ agg, float* __restrict__ deg) {
    int i = blockIdx.x * blockDim.x + threadIdx.x;
    if (i < NN * DHID) agg[i] = 0.0f;
    if (i < NN)        deg[i] = 1.0f;
}

// ---------------- in-degree over edge targets ----------------
__global__ void k_deg(const int* __restrict__ col, float* __restrict__ deg) {
    int e = blockIdx.x * blockDim.x + threadIdx.x;
    if (e < NE) atomicAdd(&deg[col[e]], 1.0f);
}

__global__ void k_dinv(const float* __restrict__ deg, float* __restrict__ dinv) {
    int n = blockIdx.x * blockDim.x + threadIdx.x;
    if (n < NN) dinv[n] = rsqrtf(fmaxf(deg[n], 1.0f));
}

// ---------------- message scatter: 16 threads / edge, float4 each ----------------
__global__ void k_scatter(const int* __restrict__ src, const int* __restrict__ dst,
                          const float* __restrict__ h0, const float* __restrict__ dinv,
                          float* __restrict__ agg) {
    int t = blockIdx.x * blockDim.x + threadIdx.x;
    int e = t >> 4;
    if (e >= NE) return;
    int sub = t & 15;
    int r = src[e], c = dst[e];
    float norm = dinv[r] * dinv[c];
    float4 v = *(const float4*)(h0 + (size_t)r * DHID + sub * 4);
    float* ap = agg + (size_t)c * DHID + sub * 4;
    atomicAdd(ap + 0, v.x * norm);
    atomicAdd(ap + 1, v.y * norm);
    atomicAdd(ap + 2, v.z * norm);
    atomicAdd(ap + 3, v.w * norm);
}

// ---------------- + self-loop + b1, ReLU, BN(eval), ReLU (in place on agg) ----------------
__global__ void k_post(const float* __restrict__ h0, const float* __restrict__ dinv,
                       const float* __restrict__ b1, const float* __restrict__ gamma,
                       const float* __restrict__ beta, const float* __restrict__ mean,
                       const float* __restrict__ var, float* __restrict__ agg) {
    int i = blockIdx.x * blockDim.x + threadIdx.x;
    if (i >= NN * DHID) return;
    int n = i >> 6, d = i & 63;
    float di = dinv[n];
    float v = agg[i] + h0[i] * di * di + b1[d];
    v = fmaxf(v, 0.0f);
    v = (v - mean[d]) * rsqrtf(var[d] + BN_EPS) * gamma[d] + beta[d];
    v = fmaxf(v, 0.0f);
    agg[i] = v;
}

// ---------------- edge head: one wave32 per edge ----------------
__global__ void k_edge(const int* __restrict__ src, const int* __restrict__ dst,
                       const float* __restrict__ h, const float* __restrict__ Wfc,
                       const float* __restrict__ bfc, float* __restrict__ out) {
    int t = blockIdx.x * blockDim.x + threadIdx.x;
    int e = t >> 5;
    if (e >= NE) return;
    int lane = threadIdx.x & 31;
    int half = lane >> 4;          // 0 -> src row, 1 -> dst row
    int seg  = lane & 15;          // 4 floats each
    int node = half ? dst[e] : src[e];
    float4 hv = *(const float4*)(h + (size_t)node * DHID + seg * 4);
    int c0 = half * 64 + seg * 4;  // concat dim base
    float o0 = hv.x * Wfc[(c0 + 0) * 2]     + hv.y * Wfc[(c0 + 1) * 2]
             + hv.z * Wfc[(c0 + 2) * 2]     + hv.w * Wfc[(c0 + 3) * 2];
    float o1 = hv.x * Wfc[(c0 + 0) * 2 + 1] + hv.y * Wfc[(c0 + 1) * 2 + 1]
             + hv.z * Wfc[(c0 + 2) * 2 + 1] + hv.w * Wfc[(c0 + 3) * 2 + 1];
#pragma unroll
    for (int off = 16; off > 0; off >>= 1) {
        o0 += __shfl_xor(o0, off, 32);
        o1 += __shfl_xor(o1, off, 32);
    }
    if (lane == 0) {
        out[(size_t)e * 2 + 0] = o0 + bfc[0];
        out[(size_t)e * 2 + 1] = o1 + bfc[1];
    }
}

extern "C" void kernel_launch(void* const* d_in, const int* in_sizes, int n_in,
                              void* d_out, int out_size, void* d_ws, size_t ws_size,
                              hipStream_t stream) {
    const float* x     = (const float*)d_in[0];
    const int*   ei    = (const int*)d_in[1];    // [2, NE] int32 (JAX x64 off)
    const float* W1    = (const float*)d_in[2];
    const float* b1    = (const float*)d_in[3];
    const float* gamma = (const float*)d_in[4];
    const float* beta  = (const float*)d_in[5];
    const float* mean  = (const float*)d_in[6];
    const float* var   = (const float*)d_in[7];
    const float* Wfc   = (const float*)d_in[8];
    const float* bfc   = (const float*)d_in[9];
    float* out = (float*)d_out;

    char* ws = (char*)d_ws;
    float*  h0   = (float*)(ws + 0);            // 25,600,000 B
    float*  agg  = (float*)(ws + 25600000);     // 25,600,000 B
    float*  deg  = (float*)(ws + 51200000);     //    400,000 B
    float*  dinv = (float*)(ws + 51600000);     //    400,000 B
    __bf16* Bp   = (__bf16*)(ws + 52000000);    //     16,384 B (32B-aligned)
    const int* srcv = ei;
    const int* dstv = ei + NE;

    k_pack_w <<<2, 256, 0, stream>>>(W1, Bp);
    k_gemm   <<<(6250 * 32 + 255) / 256, 256, 0, stream>>>(x, Bp, h0);
    k_init   <<<(NN * DHID + 255) / 256, 256, 0, stream>>>(agg, deg);
    k_deg    <<<(NE + 255) / 256, 256, 0, stream>>>(dstv, deg);
    k_dinv   <<<(NN + 255) / 256, 256, 0, stream>>>(deg, dinv);
    k_scatter<<<(NE * 16 + 255) / 256, 256, 0, stream>>>(srcv, dstv, h0, dinv, agg);
    k_post   <<<(NN * DHID + 255) / 256, 256, 0, stream>>>(h0, dinv, b1, gamma, beta, mean, var, agg);
    k_edge   <<<(NE * 32 + 255) / 256, 256, 0, stream>>>(srcv, dstv, agg, Wfc, bfc, out);
}
